// SeisDiT_29678224015619
// MI455X (gfx1250) — matrix operand order repaired and checked by reading, not compiled, not run on hardware
//
#include <hip/hip_runtime.h>
#include <hip/hip_bf16.h>

// ---------------------------------------------------------------------------
// SeisDiT attention block for MI455X (gfx1250, wave32, WMMA + async/TDM).
//   [cvt]   x, w_qkv, w_proj  fp32 -> bf16
//   [gemm]  qkv_f32 = Xbf16 @ Wqkv^T      (WMMA, async-to-LDS double buffered)
//   [norm]  RMSNorm(q,k) + exp-RoPE  -> Q/K/V bf16 [bw][head][h][64]
//   [attn]  flash attention per (bw,head) (WMMA, TDM tensor_load_to_lds tiles)
//   [gemm]  out = O @ Wproj^T + b
// ---------------------------------------------------------------------------

#ifndef USE_TDM
#define USE_TDM 1
#endif

#if defined(__has_builtin)
#  if __has_builtin(__builtin_amdgcn_global_load_async_to_lds_b128)
#    define HAVE_ASYNC_B128 1
#  else
#    define HAVE_ASYNC_B128 0
#  endif
#  if __has_builtin(__builtin_amdgcn_s_wait_asynccnt)
#    define HAVE_WAIT_ASYNC 1
#  else
#    define HAVE_WAIT_ASYNC 0
#  endif
#  if __has_builtin(__builtin_amdgcn_s_wait_tensorcnt)
#    define HAVE_WAIT_TENSOR 1
#  else
#    define HAVE_WAIT_TENSOR 0
#  endif
#else
#  define HAVE_ASYNC_B128 0
#  define HAVE_WAIT_ASYNC 0
#  define HAVE_WAIT_TENSOR 0
#endif

typedef __attribute__((ext_vector_type(16))) __bf16 v16bf;
typedef __attribute__((ext_vector_type(8)))  __bf16 v8bf;
typedef __attribute__((ext_vector_type(4)))  __bf16 v4bf;
typedef __attribute__((ext_vector_type(8)))  float  v8f;
typedef __attribute__((ext_vector_type(4)))  int      v4i;
typedef __attribute__((ext_vector_type(4)))  unsigned v4u;
typedef __attribute__((ext_vector_type(8)))  unsigned v8u;

// builtin signature (from compiler diagnostics):
//   (v4i* src_generic, v4i addrspace(3)* dst_lds, int, int)
#define AS_GLOBAL(p) ((v4i*)(p))
#define AS_LOCAL(p)  ((__attribute__((address_space(3))) v4i*)(p))

__device__ __forceinline__ void wait_async0() {
#if HAVE_WAIT_ASYNC
    __builtin_amdgcn_s_wait_asynccnt(0);
#else
    asm volatile("s_wait_asynccnt 0x0" ::: "memory");
#endif
}

__device__ __forceinline__ void wait_tensor0() {
#if HAVE_WAIT_TENSOR
    __builtin_amdgcn_s_wait_tensorcnt(0);
#else
    asm volatile("s_wait_tensorcnt 0x0" ::: "memory");
#endif
}

#if USE_TDM
// TDM 2D tile load: rows x 64 bf16 elements, row stride 64 elements, into LDS.
// Descriptor per CDNA5 ISA (D# groups 0/1; groups 2/3 NULL).
__device__ __forceinline__ void tdm_load_tile64(unsigned lds_off, const void* gptr,
                                                unsigned rows) {
    const unsigned long long ga = (unsigned long long)gptr;
    v4u g0;
    g0[0] = 1u;                                         // count=1, user mode
    g0[1] = lds_off;                                    // lds_addr
    g0[2] = (unsigned)(ga & 0xFFFFFFFFu);               // global_addr[31:0]
    g0[3] = (unsigned)((ga >> 32) & 0x01FFFFFFu)        // global_addr[56:32]
          | (2u << 30);                                 // type = 2 ("image")
    v8u g1;
    g1[0] = 0x00010000u;           // wg_mask=0, data_size=1 (2 bytes)
    g1[1] = 64u << 16;             // tensor_dim0 = 64 (low16)
    g1[2] = rows << 16;            // tensor_dim0 hi16=0 | tensor_dim1 low16
    g1[3] = 64u << 16;             // tensor_dim1 hi16=0 | tile_dim0 = 64
    g1[4] = rows;                  // tile_dim1 = rows, tile_dim2 = 0
    g1[5] = 64u;                   // tensor_dim0_stride = 64 (low32)
    g1[6] = 0u;                    // stride hi | tensor_dim1_stride lo
    g1[7] = 0u;
    asm volatile("tensor_load_to_lds %0, %1" :: "s"(g0), "s"(g1) : "memory");
}
#endif

__device__ __forceinline__ v8f zero_v8f() {
    v8f z;
#pragma unroll
    for (int i = 0; i < 8; ++i) z[i] = 0.0f;
    return z;
}

__device__ __forceinline__ v8f wmma_bf16(v16bf a, v16bf b, v8f c) {
    return __builtin_amdgcn_wmma_f32_16x16x32_bf16(
        false, a, false, b, (short)0, c, false, false);
}

// A fragment (16x32 bf16) from row-major [16][ld].
__device__ __forceinline__ v16bf load_a_frag(const __bf16* base, int ld, int lane) {
    const int m    = lane & 15;
    const int koff = (lane >> 4) << 4;
    v8bf lo = *(const v8bf*)(base + m * ld + koff);
    v8bf hi = *(const v8bf*)(base + m * ld + 8 + koff);
    v16bf a;
#pragma unroll
    for (int j = 0; j < 8; ++j) { a[j] = lo[j]; a[8 + j] = hi[j]; }
    return a;
}

// B fragment from B^T row-major (rows=n, cols=k contiguous).
__device__ __forceinline__ v16bf load_b_frag_rowk(const __bf16* base, int ld, int lane) {
    const int n    = lane & 15;
    const int koff = (lane >> 4) << 4;
    return *(const v16bf*)(base + n * ld + koff);
}

// B fragment from B row-major (rows=k, cols=n strided).
__device__ __forceinline__ v16bf load_b_frag_coln(const __bf16* base, int ld, int lane) {
    const int n  = lane & 15;
    const int kb = (lane >> 4) << 4;
    v16bf b;
#pragma unroll
    for (int j = 0; j < 16; ++j) b[j] = base[(kb + j) * ld + n];
    return b;
}

__device__ __forceinline__ float half16_max(float v) {
    v = fmaxf(v, __shfl_xor(v, 1));
    v = fmaxf(v, __shfl_xor(v, 2));
    v = fmaxf(v, __shfl_xor(v, 4));
    v = fmaxf(v, __shfl_xor(v, 8));
    return v;
}
__device__ __forceinline__ float half16_sum(float v) {
    v += __shfl_xor(v, 1);
    v += __shfl_xor(v, 2);
    v += __shfl_xor(v, 4);
    v += __shfl_xor(v, 8);
    return v;
}
__device__ __forceinline__ float wave32_sum(float v) {
    v += __shfl_xor(v, 1);
    v += __shfl_xor(v, 2);
    v += __shfl_xor(v, 4);
    v += __shfl_xor(v, 8);
    v += __shfl_xor(v, 16);
    return v;
}

// ---------------------------------------------------------------------------
// fp32 -> bf16 conversion
// ---------------------------------------------------------------------------
__global__ void cvt_f32_bf16_kernel(const float* __restrict__ in,
                                    __bf16* __restrict__ out, int n4) {
    int i = blockIdx.x * blockDim.x + threadIdx.x;
    if (i < n4) {
        float4 v = ((const float4*)in)[i];
        v4bf o;
        o[0] = (__bf16)v.x; o[1] = (__bf16)v.y;
        o[2] = (__bf16)v.z; o[3] = (__bf16)v.w;
        ((v4bf*)out)[i] = o;
    }
}

// ---------------------------------------------------------------------------
// GEMM: C[M,N] (f32) = A[M,K] bf16 @ B[N,K]^T bf16 (+bias)
// 256 thr / 8 waves, tile 128x128, BK=32, wave tile 64x32.
// Async-to-LDS double-buffered staging when available.
// ---------------------------------------------------------------------------
__global__ __launch_bounds__(256) void gemm_bf16_wmma_kernel(
    const __bf16* __restrict__ A, const __bf16* __restrict__ B,
    float* __restrict__ C, const float* __restrict__ bias,
    int M, int N, int K) {
    const int tid  = threadIdx.x;
    const int lane = tid & 31;
    const int wid  = tid >> 5;
    const int wm   = wid & 1;
    const int wn   = wid >> 1;
    const long m0  = (long)blockIdx.x * 128;
    const long n0  = (long)blockIdx.y * 128;

    v8f acc[4][2];
#pragma unroll
    for (int i = 0; i < 4; ++i)
#pragma unroll
        for (int j = 0; j < 2; ++j) acc[i][j] = zero_v8f();

#if HAVE_ASYNC_B128
    __shared__ __align__(32) __bf16 sA[2][128 * 32];
    __shared__ __align__(32) __bf16 sB[2][128 * 32];

    auto stage_async = [&](int buf, int k0) {
#pragma unroll
        for (int s = 0; s < 2; ++s) {
            const int slot = tid + s * 256;
            const int r = slot >> 2, g = slot & 3;
            __builtin_amdgcn_global_load_async_to_lds_b128(
                AS_GLOBAL(&A[(m0 + r) * K + k0 + g * 8]),
                AS_LOCAL(&sA[buf][r * 32 + g * 8]), 0, 0);
            __builtin_amdgcn_global_load_async_to_lds_b128(
                AS_GLOBAL(&B[(n0 + r) * K + k0 + g * 8]),
                AS_LOCAL(&sB[buf][r * 32 + g * 8]), 0, 0);
        }
    };

    stage_async(0, 0);
    wait_async0();
    __syncthreads();

    int buf = 0;
    for (int k0 = 0; k0 < K; k0 += 32) {
        if (k0 + 32 < K) stage_async(buf ^ 1, k0 + 32);   // overlap copy w/ wmma

        v16bf bfr[2];
#pragma unroll
        for (int j = 0; j < 2; ++j)
            bfr[j] = load_b_frag_rowk(&sB[buf][(wn * 32 + j * 16) * 32], 32, lane);
#pragma unroll
        for (int i = 0; i < 4; ++i) {
            v16bf afr = load_a_frag(&sA[buf][(wm * 64 + i * 16) * 32], 32, lane);
#pragma unroll
            for (int j = 0; j < 2; ++j) acc[i][j] = wmma_bf16(afr, bfr[j], acc[i][j]);
        }
        wait_async0();
        __syncthreads();
        buf ^= 1;
    }
#else
    __shared__ __align__(32) __bf16 sA[128 * 32];
    __shared__ __align__(32) __bf16 sB[128 * 32];

    for (int k0 = 0; k0 < K; k0 += 32) {
        if (k0 + 32 < K) {
            __builtin_prefetch(&A[(m0 + (tid >> 1)) * K + k0 + 32], 0, 1);
            __builtin_prefetch(&B[(n0 + (tid >> 1)) * K + k0 + 32], 0, 1);
        }
#pragma unroll
        for (int s = 0; s < 2; ++s) {
            const int slot = tid + s * 256;
            const int r = slot >> 2, g = slot & 3;
            *(v8bf*)&sA[r * 32 + g * 8] = *(const v8bf*)&A[(m0 + r) * K + k0 + g * 8];
            *(v8bf*)&sB[r * 32 + g * 8] = *(const v8bf*)&B[(n0 + r) * K + k0 + g * 8];
        }
        __syncthreads();

        v16bf bfr[2];
#pragma unroll
        for (int j = 0; j < 2; ++j)
            bfr[j] = load_b_frag_rowk(&sB[(wn * 32 + j * 16) * 32], 32, lane);
#pragma unroll
        for (int i = 0; i < 4; ++i) {
            v16bf afr = load_a_frag(&sA[(wm * 64 + i * 16) * 32], 32, lane);
#pragma unroll
            for (int j = 0; j < 2; ++j) acc[i][j] = wmma_bf16(afr, bfr[j], acc[i][j]);
        }
        __syncthreads();
    }
#endif

    // epilogue
#pragma unroll
    for (int i = 0; i < 4; ++i) {
#pragma unroll
        for (int j = 0; j < 2; ++j) {
            const int rb  = wm * 64 + i * 16 + ((lane >> 4) << 3);
            const int col = wn * 32 + j * 16 + (lane & 15);
            const float bv = bias ? bias[n0 + col] : 0.0f;
#pragma unroll
            for (int r = 0; r < 8; ++r)
                C[(m0 + rb + r) * N + n0 + col] = acc[i][j][r] + bv;
        }
    }
}

// ---------------------------------------------------------------------------
// RMSNorm(q,k) + exp-RoPE + V passthrough, fp32 -> bf16.
// ---------------------------------------------------------------------------
__global__ __launch_bounds__(256) void norm_rope_kernel(
    const float* __restrict__ qkv,
    const float* __restrict__ qw, const float* __restrict__ kw,
    __bf16* __restrict__ Q, __bf16* __restrict__ Kb, __bf16* __restrict__ Vb) {
    const int m    = blockIdx.x;
    const int lane = threadIdx.x & 31;
    const int head = threadIdx.x >> 5;
    const int h    = m >> 8;
    const int bw   = m & 255;

    const float pos  = (float)h * (1.0f / 255.0f);
    const float freq = __expf(-12.0f + 12.0f * (float)lane * (1.0f / 31.0f));
    const float ang  = pos * freq * 256.0f;
    float sn, cs;
    __sincosf(ang, &sn, &cs);

    const long obase = (((long)bw * 8 + head) * 256 + h) * 64 + lane * 2;
    const long ibase = (long)m * 1536 + head * 64 + lane * 2;

#pragma unroll
    for (int part = 0; part < 3; ++part) {
        float2 v = *(const float2*)(qkv + ibase + part * 512);
        if (part < 2) {
            float ss  = wave32_sum(v.x * v.x + v.y * v.y);
            float inv = rsqrtf(ss * (1.0f / 64.0f) + 1e-6f);
            const float* w = (part == 0) ? qw : kw;
            float e = v.x * inv * w[lane * 2];
            float o = v.y * inv * w[lane * 2 + 1];
            v.x = e * cs - o * sn;
            v.y = e * sn + o * cs;
        }
        __bf16* dst = (part == 0) ? Q : (part == 1) ? Kb : Vb;
        dst[obase]     = (__bf16)v.x;
        dst[obase + 1] = (__bf16)v.y;
    }
}

// ---------------------------------------------------------------------------
// Flash attention per (qblock, bw, head); K/V/Q tiles staged via TDM.
// ---------------------------------------------------------------------------
__global__ __launch_bounds__(256) void attn_flash_kernel(
    const __bf16* __restrict__ Q, const __bf16* __restrict__ Kg,
    const __bf16* __restrict__ Vg, __bf16* __restrict__ O) {
    const int qb   = blockIdx.x;
    const int bw   = blockIdx.y;
    const int head = blockIdx.z;
    const int tid  = threadIdx.x;
    const int lane = tid & 31;
    const int wid  = tid >> 5;
    const int qr0  = wid * 16;
    const float scale = 0.125f;

    __shared__ __align__(32) __bf16 sQ[128 * 64];
    __shared__ __align__(32) __bf16 sK[64 * 64];
    __shared__ __align__(32) __bf16 sV[64 * 64];
    __shared__ __align__(32) __bf16 sP[128 * 64];

    const long base = ((long)bw * 8 + head) * 256 * 64;

#if USE_TDM
    if (wid == 0) {
        tdm_load_tile64((unsigned)(unsigned long long)(void*)sQ,
                        Q + base + (long)qb * 128 * 64, 128u);
    }
#else
#pragma unroll
    for (int s = 0; s < 4; ++s) {
        const int slot = tid + s * 256;
        const int r = slot >> 3, g = slot & 7;
        *(v8bf*)&sQ[r * 64 + g * 8] =
            *(const v8bf*)&Q[base + (long)(qb * 128 + r) * 64 + g * 8];
    }
#endif

    v8f oacc[4];
#pragma unroll
    for (int nt = 0; nt < 4; ++nt) oacc[nt] = zero_v8f();
    float mprev[8], lsum[8];
#pragma unroll
    for (int r = 0; r < 8; ++r) { mprev[r] = -1e30f; lsum[r] = 0.0f; }

    for (int c = 0; c < 4; ++c) {
        __syncthreads();                 // previous chunk fully consumed
#if USE_TDM
        if (wid == 0) {
            tdm_load_tile64((unsigned)(unsigned long long)(void*)sK,
                            Kg + base + (long)c * 64 * 64, 64u);
            tdm_load_tile64((unsigned)(unsigned long long)(void*)sV,
                            Vg + base + (long)c * 64 * 64, 64u);
            wait_tensor0();              // also covers the sQ TDM on c==0
        }
#else
#pragma unroll
        for (int s = 0; s < 2; ++s) {
            const int slot = tid + s * 256;
            const int r = slot >> 3, g = slot & 7;
            *(v8bf*)&sK[r * 64 + g * 8] =
                *(const v8bf*)&Kg[base + (long)(c * 64 + r) * 64 + g * 8];
            *(v8bf*)&sV[r * 64 + g * 8] =
                *(const v8bf*)&Vg[base + (long)(c * 64 + r) * 64 + g * 8];
        }
#endif
        __syncthreads();

        // S(16x64) = Q @ K^T
        v8f sacc[4];
#pragma unroll
        for (int nt = 0; nt < 4; ++nt) sacc[nt] = zero_v8f();
#pragma unroll
        for (int ks = 0; ks < 2; ++ks) {
            v16bf aq = load_a_frag(&sQ[qr0 * 64 + ks * 32], 64, lane);
#pragma unroll
            for (int nt = 0; nt < 4; ++nt) {
                v16bf bk = load_b_frag_rowk(&sK[(nt * 16) * 64 + ks * 32], 64, lane);
                sacc[nt] = wmma_bf16(aq, bk, sacc[nt]);
            }
        }
#pragma unroll
        for (int nt = 0; nt < 4; ++nt)
#pragma unroll
            for (int r = 0; r < 8; ++r) sacc[nt][r] *= scale;

        // online softmax (rows live in one 16-lane half)
#pragma unroll
        for (int r = 0; r < 8; ++r) {
            float lm = fmaxf(fmaxf(sacc[0][r], sacc[1][r]),
                             fmaxf(sacc[2][r], sacc[3][r]));
            lm = half16_max(lm);
            const float mnew  = fmaxf(mprev[r], lm);
            const float alpha = __expf(mprev[r] - mnew);
            lsum[r] *= alpha;
#pragma unroll
            for (int nt = 0; nt < 4; ++nt) oacc[nt][r] *= alpha;
            float psum = 0.0f;
#pragma unroll
            for (int nt = 0; nt < 4; ++nt) {
                const float p = __expf(sacc[nt][r] - mnew);
                sacc[nt][r] = p;
                psum += p;
            }
            lsum[r] += half16_sum(psum);
            mprev[r] = mnew;
        }

        // P -> bf16 in wave-private LDS rows
#pragma unroll
        for (int nt = 0; nt < 4; ++nt) {
            const int col = nt * 16 + (lane & 15);
            const int ro  = (lane >> 4) << 3;
#pragma unroll
            for (int r = 0; r < 8; ++r)
                sP[(qr0 + ro + r) * 64 + col] = (__bf16)sacc[nt][r];
        }

        // O += P @ V
#pragma unroll
        for (int ks = 0; ks < 2; ++ks) {
            v16bf ap = load_a_frag(&sP[qr0 * 64 + ks * 32], 64, lane);
#pragma unroll
            for (int nt = 0; nt < 4; ++nt) {
                v16bf bv = load_b_frag_coln(&sV[(ks * 32) * 64 + nt * 16], 64, lane);
                oacc[nt] = wmma_bf16(ap, bv, oacc[nt]);
            }
        }
    }

    // finalize + store (token-major for output GEMM)
#pragma unroll
    for (int nt = 0; nt < 4; ++nt) {
        const int d  = nt * 16 + (lane & 15);
        const int ro = (lane >> 4) << 3;
#pragma unroll
        for (int r = 0; r < 8; ++r) {
            const int hrow = qb * 128 + qr0 + ro + r;
            const float ov = oacc[nt][r] / lsum[r];
            O[((long)hrow * 256 + bw) * 512 + head * 64 + d] = (__bf16)ov;
        }
    }
}

// ---------------------------------------------------------------------------
// launch
// ---------------------------------------------------------------------------
extern "C" void kernel_launch(void* const* d_in, const int* in_sizes, int n_in,
                              void* d_out, int out_size, void* d_ws, size_t ws_size,
                              hipStream_t stream) {
    const float* x      = (const float*)d_in[0];
    const float* w_qkv  = (const float*)d_in[1];
    const float* qnw    = (const float*)d_in[2];
    const float* knw    = (const float*)d_in[3];
    const float* w_proj = (const float*)d_in[4];
    const float* b_proj = (const float*)d_in[5];
    float* out = (float*)d_out;

    char* ws = (char*)d_ws;
    constexpr size_t OFF_XB    = 0;
    constexpr size_t OFF_WQKV  = OFF_XB    + 67108864ULL;
    constexpr size_t OFF_WPROJ = OFF_WQKV  + 1572864ULL;
    constexpr size_t OFF_QKV   = OFF_WPROJ + 524288ULL;
    constexpr size_t OFF_Q     = OFF_QKV   + 402653184ULL;
    constexpr size_t OFF_K     = OFF_Q     + 67108864ULL;
    constexpr size_t OFF_V     = OFF_K     + 67108864ULL;
    constexpr size_t OFF_O     = OFF_V     + 67108864ULL;

    __bf16* Xb    = (__bf16*)(ws + OFF_XB);
    __bf16* Wqkvb = (__bf16*)(ws + OFF_WQKV);
    __bf16* Wprjb = (__bf16*)(ws + OFF_WPROJ);
    float*  qkv   = (float*)(ws + OFF_QKV);
    __bf16* Qb    = (__bf16*)(ws + OFF_Q);
    __bf16* Kb    = (__bf16*)(ws + OFF_K);
    __bf16* Vb    = (__bf16*)(ws + OFF_V);
    __bf16* Ob    = (__bf16*)(ws + OFF_O);

    {
        int n4 = 33554432 / 4;
        cvt_f32_bf16_kernel<<<(n4 + 255) / 256, 256, 0, stream>>>(x, Xb, n4);
        n4 = 786432 / 4;
        cvt_f32_bf16_kernel<<<(n4 + 255) / 256, 256, 0, stream>>>(w_qkv, Wqkvb, n4);
        n4 = 262144 / 4;
        cvt_f32_bf16_kernel<<<(n4 + 255) / 256, 256, 0, stream>>>(w_proj, Wprjb, n4);
    }

    gemm_bf16_wmma_kernel<<<dim3(512, 12), 256, 0, stream>>>(
        Xb, Wqkvb, qkv, nullptr, 65536, 1536, 512);

    norm_rope_kernel<<<65536, 256, 0, stream>>>(qkv, qnw, knw, Qb, Kb, Vb);

    attn_flash_kernel<<<dim3(2, 256, 8), 256, 0, stream>>>(Qb, Kb, Vb, Ob);

    gemm_bf16_wmma_kernel<<<dim3(512, 4), 256, 0, stream>>>(
        Ob, Wprjb, out, b_proj, 65536, 512, 512);
}
